// ReLuGRUCell_51187420234361
// MI455X (gfx1250) — matrix operand order repaired and checked by reading, not compiled, not run on hardware
//
#include <hip/hip_runtime.h>
#include <hip/hip_bf16.h>

#define BATCH 16384
#define DIN   1024
#define HDIM  1024

typedef __attribute__((ext_vector_type(16))) __bf16 v16bf;
typedef __attribute__((ext_vector_type(8)))  __bf16 v8bf;
typedef __attribute__((ext_vector_type(8)))  float  v8f;

__device__ __forceinline__ float sigmoidf_(float x) {
  return 1.0f / (1.0f + __expf(-x));
}

__device__ __forceinline__ v8f wmma_bf16(v16bf a, v16bf b, v8f c) {
  // (neg_a, A, neg_b, B, c_mod, C, reuse_a, reuse_b)
  return __builtin_amdgcn_wmma_f32_16x16x32_bf16(false, a, false, b, (short)0, c,
                                                 false, false);
}

// CDNA5 async global->LDS copy, 16B per lane, tracked by ASYNCcnt.
// LDS address comes from the vdst VGPR; low 32 bits of a generic LDS pointer are the
// LDS byte address (aperture lives in addr[63:32]).
__device__ __forceinline__ void async_copy_b128(unsigned lds_addr,
                                                unsigned long long gaddr) {
  asm volatile("global_load_async_to_lds_b128 %0, %1, off"
               :: "v"(lds_addr), "v"(gaddr) : "memory");
}
__device__ __forceinline__ void wait_async0() {
  asm volatile("s_wait_asynccnt 0" ::: "memory");
}

// ---- fp32 -> bf16 bulk converter (memory-bound pre-pass) -------------------
__global__ __launch_bounds__(256) void cvt_f32_to_bf16(
    const float* __restrict__ src, unsigned short* __restrict__ dst, int n8)
{
  int i = blockIdx.x * blockDim.x + threadIdx.x;
  if (i >= n8) return;
  const float4* p = (const float4*)src + (size_t)i * 2;
  float4 x = p[0];
  float4 y = p[1];
  v8bf t;
  t[0] = (__bf16)x.x; t[1] = (__bf16)x.y; t[2] = (__bf16)x.z; t[3] = (__bf16)x.w;
  t[4] = (__bf16)y.x; t[5] = (__bf16)y.y; t[6] = (__bf16)y.z; t[7] = (__bf16)y.w;
  ((v8bf*)dst)[i] = t;
}

// A fragment 16x32 bf16 per ISA layout: lanes 0-15 rows M, K={0..7,16..23};
// lanes 16-31 same rows, K={8..15,24..31}. tile is [rows][32] row-major.
__device__ __forceinline__ v16bf frag_a(const __bf16* tile, int mbase, int lane) {
  int r  = mbase + (lane & 15);
  int k0 = (lane < 16) ? 0 : 8;
  const __bf16* p = tile + r * 32 + k0;
  union { v16bf v; v8bf h[2]; } u;
  u.h[0] = *(const v8bf*)(p);        // K = k0 .. k0+7
  u.h[1] = *(const v8bf*)(p + 16);   // K = k0+16 .. k0+23
  return u.v;
}

// B fragment 32x16 bf16 (B[k][n] = W[nbase+n][k]): lane%16 = column N,
// lanes 0-15 hold K={0..7,16..23}, lanes 16-31 hold K={8..15,24..31}.
__device__ __forceinline__ v16bf frag_b(const __bf16* tile, int nbase, int lane) {
  int r  = nbase + (lane & 15);
  int k0 = (lane < 16) ? 0 : 8;
  const __bf16* p = tile + r * 32 + k0;
  union { v16bf v; v8bf h[2]; } u;
  u.h[0] = *(const v8bf*)(p);
  u.h[1] = *(const v8bf*)(p + 16);
  return u.v;
}

// -------- Phase 1: 5 independent GEMMs fused + gate epilogue ----------------
// WG tile: 64 (M) x 128 (N). 8 waves; wave wv owns the full-M 64x16 strip at
// N = N0 + wv*16. Per K-step: 8 A-frags + 5 B-frags -> 20 WMMAs.
__global__ __launch_bounds__(256) void gru_phase1(
    const unsigned short* __restrict__ in_bf_u16,
    const unsigned short* __restrict__ hx_bf_u16,
    const unsigned short* __restrict__ wih_bf_u16,
    const unsigned short* __restrict__ whh_bf_u16,
    const float* __restrict__ hx, const float* __restrict__ bias,
    float* __restrict__ zbuf, float* __restrict__ npart,
    unsigned short* __restrict__ rhbuf_u16)
{
  // Buffer layout (bf16 elems): A_in[0,2048) A_hx[2048,4096) W[g] 4096+g*4096
  __shared__ __bf16 smem[2][24576];   // 2 x 48KB

  const __bf16* in_bf  = (const __bf16*)in_bf_u16;
  const __bf16* hx_bf  = (const __bf16*)hx_bf_u16;
  const __bf16* wih_bf = (const __bf16*)wih_bf_u16;
  const __bf16* whh_bf = (const __bf16*)whh_bf_u16;
  __bf16* rhbuf = (__bf16*)rhbuf_u16;

  const int tid  = threadIdx.x;
  const int lane = tid & 31;
  const int wv   = tid >> 5;        // 0..7 -> N strip
  const int M0   = blockIdx.y << 6; // 64 rows
  const int N0   = blockIdx.x << 7; // 128 cols

  // staging role: thread covers (row r, cols c..c+7) of each 32-wide K slice
  const int r = tid >> 2;           // 0..63
  const int c = (tid & 3) << 3;     // 0,8,16,24

  // 12 async copies per K-step: A_in, A_hx, then (g,half) for 5 W tiles.
  unsigned long long ga[12];
  unsigned           ldoff[12];
  ga[0]    = (unsigned long long)(uintptr_t)(in_bf + (size_t)(M0 + r) * DIN + c);
  ldoff[0] = (unsigned)((r * 32 + c) * 2);
  ga[1]    = (unsigned long long)(uintptr_t)(hx_bf + (size_t)(M0 + r) * HDIM + c);
  ldoff[1] = (unsigned)(4096 + (r * 32 + c) * 2);
  {
    int q = 2;
#pragma unroll
    for (int g = 0; g < 5; ++g) {
      const __bf16* wbase = (g < 3) ? (wih_bf + (size_t)g * HDIM * DIN)
                                    : (whh_bf + (size_t)(g - 3) * HDIM * HDIM);
#pragma unroll
      for (int half = 0; half < 2; ++half) {
        int wr = half * 64 + r;   // 0..127
        ga[q]    = (unsigned long long)(uintptr_t)(wbase + (size_t)(N0 + wr) * 1024 + c);
        ldoff[q] = (unsigned)(8192 + g * 8192 + (wr * 32 + c) * 2);
        ++q;
      }
    }
  }
  const unsigned lds_base = (unsigned)(uintptr_t)(&smem[0][0]);

  // issue async fills of buffer `buf`, then advance sources by 32 elems (64B)
  auto issue = [&](int buf) {
    const unsigned b = lds_base + (unsigned)buf * 49152u;
#pragma unroll
    for (int t = 0; t < 12; ++t) {
      async_copy_b128(b + ldoff[t], ga[t]);
      ga[t] += 64;   // next K slice
    }
  };

  v8f acc[5][4] = {};

  issue(0);
  wait_async0();
  __syncthreads();

  const int NSTEP = DIN / 32;   // 32
  for (int step = 0; step < NSTEP; ++step) {
    const int cur = step & 1;
    if (step + 1 < NSTEP) issue(cur ^ 1);

    const __bf16* sm = smem[cur];
    v16bf a_in[4], a_hx[4], bf[5];
#pragma unroll
    for (int m = 0; m < 4; ++m) {
      a_in[m] = frag_a(sm,        m * 16, lane);
      a_hx[m] = frag_a(sm + 2048, m * 16, lane);
    }
#pragma unroll
    for (int g = 0; g < 5; ++g)
      bf[g] = frag_b(sm + 4096 + g * 4096, wv * 16, lane);
#pragma unroll
    for (int g = 0; g < 5; ++g) {
#pragma unroll
      for (int m = 0; m < 4; ++m)
        acc[g][m] = wmma_bf16((g < 3) ? a_in[m] : a_hx[m], bf[g], acc[g][m]);
    }

    wait_async0();
    __syncthreads();
  }

  // Epilogue. C/D layout: VGPR i -> lanes 0-15 M=i, lanes 16-31 M=i+8; N=lane%16
  const int col = N0 + wv * 16 + (lane & 15);
  const float b0 = bias[col];
  const float b1 = bias[HDIM + col];
  const float b2 = bias[2 * HDIM + col];
#pragma unroll
  for (int m = 0; m < 4; ++m) {
#pragma unroll
    for (int i = 0; i < 8; ++i) {
      const int row  = M0 + m * 16 + ((lane < 16) ? i : i + 8);
      const size_t idx = (size_t)row * HDIM + col;
      const float hv = hx[idx];
      const float z  = sigmoidf_(acc[0][m][i] + acc[3][m][i] + b0);
      const float rr = sigmoidf_(acc[1][m][i] + acc[4][m][i] + b1);
      zbuf[idx]  = z;
      npart[idx] = acc[2][m][i] + b2;
      rhbuf[idx] = (__bf16)(rr * hv);
    }
  }
}

// -------- Phase 2: gh2 = (r*hx) @ Whh2^T, fused output epilogue -------------
// Same geometry: WG 64x128, wave owns 64x16 strip. A=rh(bf16), B=Whh2(bf16).
__global__ __launch_bounds__(256) void gru_phase2(
    const unsigned short* __restrict__ rhbuf_u16,
    const unsigned short* __restrict__ whh2_bf_u16,
    const float* __restrict__ zbuf, const float* __restrict__ npart,
    const float* __restrict__ hx, float* __restrict__ out)
{
  // Buffer layout (elems): A[0,2048) B[2048,6144)
  __shared__ __bf16 smem[2][6144];   // 2 x 12KB

  const __bf16* rhb  = (const __bf16*)rhbuf_u16;
  const __bf16* whh2 = (const __bf16*)whh2_bf_u16;

  const int tid  = threadIdx.x;
  const int lane = tid & 31;
  const int wv   = tid >> 5;
  const int M0   = blockIdx.y << 6;
  const int N0   = blockIdx.x << 7;

  const int r = tid >> 2;
  const int c = (tid & 3) << 3;

  unsigned long long ga[3];
  unsigned           ldoff[3];
  ga[0]    = (unsigned long long)(uintptr_t)(rhb + (size_t)(M0 + r) * HDIM + c);
  ldoff[0] = (unsigned)((r * 32 + c) * 2);
#pragma unroll
  for (int half = 0; half < 2; ++half) {
    int wr = half * 64 + r;
    ga[1 + half]    = (unsigned long long)(uintptr_t)(whh2 + (size_t)(N0 + wr) * HDIM + c);
    ldoff[1 + half] = (unsigned)(4096 + (wr * 32 + c) * 2);
  }
  const unsigned lds_base = (unsigned)(uintptr_t)(&smem[0][0]);

  auto issue = [&](int buf) {
    const unsigned b = lds_base + (unsigned)buf * 12288u;
#pragma unroll
    for (int t = 0; t < 3; ++t) {
      async_copy_b128(b + ldoff[t], ga[t]);
      ga[t] += 64;
    }
  };

  v8f acc[4] = {};

  issue(0);
  wait_async0();
  __syncthreads();

  const int NSTEP = HDIM / 32;
  for (int step = 0; step < NSTEP; ++step) {
    const int cur = step & 1;
    if (step + 1 < NSTEP) issue(cur ^ 1);

    const __bf16* sm = smem[cur];
    v16bf bfr = frag_b(sm + 2048, wv * 16, lane);
#pragma unroll
    for (int m = 0; m < 4; ++m)
      acc[m] = wmma_bf16(frag_a(sm, m * 16, lane), bfr, acc[m]);

    wait_async0();
    __syncthreads();
  }

  const int col = N0 + wv * 16 + (lane & 15);
#pragma unroll
  for (int m = 0; m < 4; ++m) {
#pragma unroll
    for (int i = 0; i < 8; ++i) {
      const int row  = M0 + m * 16 + ((lane < 16) ? i : i + 8);
      const size_t idx = (size_t)row * HDIM + col;
      const float n = fmaxf(npart[idx] + acc[m][i], 0.0f);   // relu
      const float z = zbuf[idx];
      out[idx] = (1.0f - z) * n + z * hx[idx];
    }
  }
}

extern "C" void kernel_launch(void* const* d_in, const int* in_sizes, int n_in,
                              void* d_out, int out_size, void* d_ws, size_t ws_size,
                              hipStream_t stream) {
  (void)in_sizes; (void)n_in; (void)out_size; (void)ws_size;

  const float* input = (const float*)d_in[0];   // [B, DIN]
  const float* hx    = (const float*)d_in[1];   // [B, H]
  const float* wih   = (const float*)d_in[2];   // [3, H, DIN]
  const float* whh   = (const float*)d_in[3];   // [3, H, H]
  const float* bias  = (const float*)d_in[4];   // [3, H]
  float* out = (float*)d_out;                   // [B, H]

  // Workspace layout (bytes)
  char* ws = (char*)d_ws;
  const size_t BH = (size_t)BATCH * HDIM;       // 16M elems
  const size_t W3 = 3ull * HDIM * DIN;          // 3M   elems
  float*          zbuf   = (float*)ws;                                  // BH f32
  float*          npart  = (float*)(ws + BH * 4);                       // BH f32
  unsigned short* rhbuf  = (unsigned short*)(ws + BH * 8);              // BH bf16
  unsigned short* in_bf  = (unsigned short*)(ws + BH * 10);             // BH bf16
  unsigned short* hx_bf  = (unsigned short*)(ws + BH * 12);             // BH bf16
  unsigned short* wih_bf = (unsigned short*)(ws + BH * 14);             // W3 bf16
  unsigned short* whh_bf = (unsigned short*)(ws + BH * 14 + W3 * 2);    // W3 bf16

  dim3 block(256);

  // Pre-pass: fp32 -> bf16 (memory-bound, done once)
  cvt_f32_to_bf16<<<dim3((int)(BH / 8 / 256)), block, 0, stream>>>(input, in_bf, (int)(BH / 8));
  cvt_f32_to_bf16<<<dim3((int)(BH / 8 / 256)), block, 0, stream>>>(hx, hx_bf, (int)(BH / 8));
  cvt_f32_to_bf16<<<dim3((int)(W3 / 8 / 256)), block, 0, stream>>>(wih, wih_bf, (int)(W3 / 8));
  cvt_f32_to_bf16<<<dim3((int)(W3 / 8 / 256)), block, 0, stream>>>(whh, whh_bf, (int)(W3 / 8));

  dim3 grid(HDIM / 128, BATCH / 64);   // (8, 256)

  gru_phase1<<<grid, block, 0, stream>>>(in_bf, hx_bf, wih_bf, whh_bf,
                                         hx, bias, zbuf, npart, rhbuf);
  gru_phase2<<<grid, block, 0, stream>>>(rhbuf,
                                         whh_bf + 2ull * HDIM * HDIM,
                                         zbuf, npart, hx, out);
}
